// StandaloneGatedGCNLayer_6347961663751
// MI455X (gfx1250) — matrix operand order repaired and checked by reading, not compiled, not run on hardware
//
#include <hip/hip_runtime.h>

#define DD 128
#define BN_EPS 1e-5f

typedef __attribute__((ext_vector_type(16))) __bf16 v16bf;
typedef __attribute__((ext_vector_type(2)))  __bf16 v2bf;
typedef __attribute__((ext_vector_type(8)))  float v8f;

union Frag { v16bf v; unsigned u[8]; };

// packs to a single v_cvt_pk_bf16_f32
__device__ __forceinline__ unsigned pk_bf16(float lo, float hi) {
  v2bf p = {(__bf16)lo, (__bf16)hi};
  return __builtin_bit_cast(unsigned, p);
}

// ---------------------------------------------------------------------------
// WMMA GEMM:  Out[M,Nout] = act(A[M,K] @ W[K,Nout] + bias)
// 256 threads = 8 wave32s; block tile 128(M) x 64(N); K stepped by 32.
// Each wave owns a 32x32 output tile: 2 A-frags x 2 B-frags -> 4 WMMAs/step.
// LDS holds bf16 PAIRS packed in uint32; W tile stored transposed so both
// fragment types are built with ds_load_2addr_b32 (no sub-word packing).
// ---------------------------------------------------------------------------
#define BM 128
#define BN 64
#define BK 32
#define LA 17   // 16 k-pairs + 1 pad (uints)
#define LW 17

__global__ __launch_bounds__(256) void wmma_gemm_bias(
    const float* __restrict__ A, const float* __restrict__ W,
    const float* __restrict__ bias, float* __restrict__ Out,
    int M, int K, int Nout, int relu)
{
  __shared__ unsigned As[BM * LA];  // [row][kpair]
  __shared__ unsigned Wt[BN * LW];  // [col][kpair] (transposed)

  const int tid  = threadIdx.x;
  const int lane = tid & 31;
  const int wave = tid >> 5;
  const int wm = wave & 3;   // 4 wave-rows * 32
  const int wn = wave >> 2;  // 2 wave-cols * 32
  const int rowBase = blockIdx.x * BM;
  const int colBase = blockIdx.y * BN;

  v8f acc[2][2];
#pragma unroll
  for (int t = 0; t < 2; ++t)
#pragma unroll
    for (int s = 0; s < 2; ++s) acc[t][s] = (v8f){0.f,0.f,0.f,0.f,0.f,0.f,0.f,0.f};

  const int mfr = lane & 15;
  const int kpA = (lane < 16) ? 0 : 4;  // A lane-group k-pair base (k 0 / 8)
  const int kpB = (lane < 16) ? 0 : 4;  // B lane-group k-pair base (k 0 / 8)

  // per-thread staging coordinates (constant across K loop)
  const int a_r[4] = { (tid + 0) >> 3, (tid + 256) >> 3,
                       (tid + 512) >> 3, (tid + 768) >> 3 };
  const int a_q = tid & 7;                      // float4 slot within row
  const int w_n4 = (tid & 15) * 4;              // 4 adjacent columns
  const int w_kp = tid >> 4;                    // k-pair 0..15

  for (int k0 = 0; k0 < K; k0 += BK) {
    // ---- issue ALL global loads first (keep them in flight together) ----
    float4 av[4];
#pragma unroll
    for (int i = 0; i < 4; ++i) {
      int grow = rowBase + a_r[i]; grow = (grow < M) ? grow : (M - 1);
      av[i] = *(const float4*)(A + (long long)grow * K + (k0 + a_q * 4));
    }
    float4 w0 = *(const float4*)(W + (long long)(k0 + 2 * w_kp + 0) * Nout + (colBase + w_n4));
    float4 w1 = *(const float4*)(W + (long long)(k0 + 2 * w_kp + 1) * Nout + (colBase + w_n4));
    if (k0 + BK < K)
      __builtin_prefetch(A + (long long)(rowBase + wave * 16) * K + (k0 + BK), 0, 1);

    // ---- convert + store to LDS ----
#pragma unroll
    for (int i = 0; i < 4; ++i) {
      As[a_r[i] * LA + a_q * 2 + 0] = pk_bf16(av[i].x, av[i].y);
      As[a_r[i] * LA + a_q * 2 + 1] = pk_bf16(av[i].z, av[i].w);
    }
    Wt[(w_n4 + 0) * LW + w_kp] = pk_bf16(w0.x, w1.x);
    Wt[(w_n4 + 1) * LW + w_kp] = pk_bf16(w0.y, w1.y);
    Wt[(w_n4 + 2) * LW + w_kp] = pk_bf16(w0.z, w1.z);
    Wt[(w_n4 + 3) * LW + w_kp] = pk_bf16(w0.w, w1.w);
    __syncthreads();

    // ---- build fragments and 4 WMMAs ----
    Frag fa[2], fb[2];
#pragma unroll
    for (int t = 0; t < 2; ++t) {
      const unsigned* base = &As[(wm * 32 + t * 16 + mfr) * LA + kpA];
#pragma unroll
      for (int j = 0; j < 8; ++j)
        fa[t].u[j] = base[((j >= 4) ? 8 : 0) + (j & 3)];
    }
#pragma unroll
    for (int s = 0; s < 2; ++s) {
      const unsigned* base = &Wt[(wn * 32 + s * 16 + mfr) * LW + kpB];
#pragma unroll
      for (int j = 0; j < 8; ++j)
        fb[s].u[j] = base[((j >= 4) ? 8 : 0) + (j & 3)];
    }
#pragma unroll
    for (int t = 0; t < 2; ++t)
#pragma unroll
      for (int s = 0; s < 2; ++s)
        acc[t][s] = __builtin_amdgcn_wmma_f32_16x16x32_bf16(
            false, fa[t].v, false, fb[s].v, (short)0, acc[t][s], false, false);
    __syncthreads();
  }

  // C/D layout: lane n = lane&15; VGPR r -> M = r (lanes 0-15) / 8+r (16-31)
  const int mb = (lane < 16) ? 0 : 8;
#pragma unroll
  for (int s = 0; s < 2; ++s) {
    const int colg = colBase + wn * 32 + s * 16 + mfr;
    const float bv = bias[colg];
#pragma unroll
    for (int t = 0; t < 2; ++t) {
#pragma unroll
      for (int r = 0; r < 8; ++r) {
        int grow = rowBase + wm * 32 + t * 16 + mb + r;
        if (grow < M) {
          float v = acc[t][s][r] + bv;
          if (relu) v = fmaxf(v, 0.f);
          Out[(long long)grow * Nout + colg] = v;
        }
      }
    }
  }
}

// ---------------------------------------------------------------------------
// Edge core: e_ij = Dx[row]+Ex[col]+Ce (in place over Ce);
// msg = sigmoid(e_ij)*Bx[col]; aggr[row] += msg (global float atomics);
// accumulate per-column sum/sum^2 of e_ij for the edge BatchNorm.
// One wave per edge, 4 features per lane (float4).
// ---------------------------------------------------------------------------
__global__ __launch_bounds__(256) void edge_gate_scatter(
    const int* __restrict__ eidx, const float* __restrict__ Dx,
    const float* __restrict__ Ex, const float* __restrict__ Bx,
    float* __restrict__ eij, float* __restrict__ aggr,
    float* __restrict__ stat_e, int E)
{
  const int gt = blockIdx.x * blockDim.x + threadIdx.x;
  const int wid = gt >> 5, lane = gt & 31;
  const int nw = (gridDim.x * blockDim.x) >> 5;
  const int f = lane * 4;
  float s[4] = {0.f, 0.f, 0.f, 0.f};
  float q[4] = {0.f, 0.f, 0.f, 0.f};

  for (int e = wid; e < E; e += nw) {
    int r = eidx[e];
    int c = eidx[E + e];
    float4 d4 = *(const float4*)(Dx + (long long)r * DD + f);
    float4 x4 = *(const float4*)(Ex + (long long)c * DD + f);
    float4 c4 = *(float4*)(eij + (long long)e * DD + f);
    float4 b4 = *(const float4*)(Bx + (long long)c * DD + f);
    float* vp = (float*)&c4;
    const float* dp = (const float*)&d4;
    const float* xp = (const float*)&x4;
    const float* bp = (const float*)&b4;
    float* ag = aggr + (long long)r * DD + f;
#pragma unroll
    for (int j = 0; j < 4; ++j) {
      float v = dp[j] + xp[j] + vp[j];
      vp[j] = v;
      s[j] += v;
      q[j] += v * v;
      float m = bp[j] / (1.f + expf(-v));
      atomicAdd(ag + j, m);
    }
    *(float4*)(eij + (long long)e * DD + f) = c4;
  }
#pragma unroll
  for (int j = 0; j < 4; ++j) {
    atomicAdd(stat_e + f + j, s[j]);
    atomicAdd(stat_e + DD + f + j, q[j]);
  }
}

// x_trans = Ax + aggr (in place over aggr) + node BN stats
__global__ __launch_bounds__(256) void node_xtrans(
    const float* __restrict__ Ax, float* __restrict__ aggr,
    float* __restrict__ stat_x, int N)
{
  const int gt = blockIdx.x * blockDim.x + threadIdx.x;
  const int wid = gt >> 5, lane = gt & 31;
  const int nw = (gridDim.x * blockDim.x) >> 5;
  const int f = lane * 4;
  float s[4] = {0.f, 0.f, 0.f, 0.f};
  float q[4] = {0.f, 0.f, 0.f, 0.f};
  for (int n = wid; n < N; n += nw) {
    float4 a4 = *(const float4*)(Ax + (long long)n * DD + f);
    float4 g4 = *(float4*)(aggr + (long long)n * DD + f);
    float* gp = (float*)&g4;
    const float* ap = (const float*)&a4;
#pragma unroll
    for (int j = 0; j < 4; ++j) {
      float v = ap[j] + gp[j];
      gp[j] = v;
      s[j] += v;
      q[j] += v * v;
    }
    *(float4*)(aggr + (long long)n * DD + f) = g4;
  }
#pragma unroll
  for (int j = 0; j < 4; ++j) {
    atomicAdd(stat_x + f + j, s[j]);
    atomicAdd(stat_x + DD + f + j, q[j]);
  }
}

// stats -> fused scale/shift:  y = h*scale + shift  == (h-m)*rstd*g + b
__global__ void bn_finalize(const float* __restrict__ stat,
                            const float* __restrict__ g,
                            const float* __restrict__ b, float inv_cnt,
                            float* __restrict__ scale, float* __restrict__ shift)
{
  int d = threadIdx.x;  // 128
  float m = stat[d] * inv_cnt;
  float var = stat[DD + d] * inv_cnt - m * m;
  float rs = rsqrtf(var + BN_EPS);
  float sc = g[d] * rs;
  scale[d] = sc;
  shift[d] = b[d] - m * sc;
}

// x_final = x + relu(bn(x_trans)); accumulate n1 stats of x_final
__global__ __launch_bounds__(256) void apply_x(
    const float* __restrict__ x, const float* __restrict__ xtrans,
    const float* __restrict__ scale, const float* __restrict__ shift,
    float* __restrict__ xfinal, float* __restrict__ stat_n1, int N)
{
  const int gt = blockIdx.x * blockDim.x + threadIdx.x;
  const int wid = gt >> 5, lane = gt & 31;
  const int nw = (gridDim.x * blockDim.x) >> 5;
  const int f = lane * 4;
  float s[4] = {0.f, 0.f, 0.f, 0.f};
  float q[4] = {0.f, 0.f, 0.f, 0.f};
  float sc[4], sh[4];
#pragma unroll
  for (int j = 0; j < 4; ++j) { sc[j] = scale[f + j]; sh[j] = shift[f + j]; }
  for (int n = wid; n < N; n += nw) {
    float4 x4 = *(const float4*)(x + (long long)n * DD + f);
    float4 t4 = *(const float4*)(xtrans + (long long)n * DD + f);
    const float* xp = (const float*)&x4;
    float* tp = (float*)&t4;
#pragma unroll
    for (int j = 0; j < 4; ++j) {
      float v = xp[j] + fmaxf(tp[j] * sc[j] + sh[j], 0.f);
      tp[j] = v;
      s[j] += v;
      q[j] += v * v;
    }
    *(float4*)(xfinal + (long long)n * DD + f) = t4;
  }
#pragma unroll
  for (int j = 0; j < 4; ++j) {
    atomicAdd(stat_n1 + f + j, s[j]);
    atomicAdd(stat_n1 + DD + f + j, q[j]);
  }
}

// e_final = edge_attr + relu(bn(e_ij))  -> straight to output
__global__ __launch_bounds__(256) void apply_e(
    const float* __restrict__ ea, const float* __restrict__ eij,
    const float* __restrict__ scale, const float* __restrict__ shift,
    float* __restrict__ efinal, int E)
{
  const int gt = blockIdx.x * blockDim.x + threadIdx.x;
  const int wid = gt >> 5, lane = gt & 31;
  const int nw = (gridDim.x * blockDim.x) >> 5;
  const int f = lane * 4;
  float sc[4], sh[4];
#pragma unroll
  for (int j = 0; j < 4; ++j) { sc[j] = scale[f + j]; sh[j] = shift[f + j]; }
  for (int e = wid; e < E; e += nw) {
    float4 a4 = *(const float4*)(ea + (long long)e * DD + f);
    float4 v4 = *(const float4*)(eij + (long long)e * DD + f);
    const float* ap = (const float*)&a4;
    float* vp = (float*)&v4;
#pragma unroll
    for (int j = 0; j < 4; ++j)
      vp[j] = ap[j] + fmaxf(vp[j] * sc[j] + sh[j], 0.f);
    *(float4*)(efinal + (long long)e * DD + f) = v4;
  }
}

// out = in*scale + shift (per column)
__global__ __launch_bounds__(256) void affine_cols(
    const float* __restrict__ in, const float* __restrict__ scale,
    const float* __restrict__ shift, float* __restrict__ out, int N)
{
  const int gt = blockIdx.x * blockDim.x + threadIdx.x;
  const int wid = gt >> 5, lane = gt & 31;
  const int nw = (gridDim.x * blockDim.x) >> 5;
  const int f = lane * 4;
  float sc[4], sh[4];
#pragma unroll
  for (int j = 0; j < 4; ++j) { sc[j] = scale[f + j]; sh[j] = shift[f + j]; }
  for (int n = wid; n < N; n += nw) {
    float4 v4 = *(const float4*)(in + (long long)n * DD + f);
    float* vp = (float*)&v4;
#pragma unroll
    for (int j = 0; j < 4; ++j) vp[j] = vp[j] * sc[j] + sh[j];
    *(float4*)(out + (long long)n * DD + f) = v4;
  }
}

// y = a + b; accumulate n2 stats of y
__global__ __launch_bounds__(256) void add_stats(
    const float* __restrict__ a, const float* __restrict__ b,
    float* __restrict__ y, float* __restrict__ stat, int N)
{
  const int gt = blockIdx.x * blockDim.x + threadIdx.x;
  const int wid = gt >> 5, lane = gt & 31;
  const int nw = (gridDim.x * blockDim.x) >> 5;
  const int f = lane * 4;
  float s[4] = {0.f, 0.f, 0.f, 0.f};
  float q[4] = {0.f, 0.f, 0.f, 0.f};
  for (int n = wid; n < N; n += nw) {
    float4 a4 = *(const float4*)(a + (long long)n * DD + f);
    float4 b4 = *(const float4*)(b + (long long)n * DD + f);
    const float* ap = (const float*)&a4;
    float* bp = (float*)&b4;
#pragma unroll
    for (int j = 0; j < 4; ++j) {
      float v = ap[j] + bp[j];
      bp[j] = v;
      s[j] += v;
      q[j] += v * v;
    }
    *(float4*)(y + (long long)n * DD + f) = b4;
  }
#pragma unroll
  for (int j = 0; j < 4; ++j) {
    atomicAdd(stat + f + j, s[j]);
    atomicAdd(stat + DD + f + j, q[j]);
  }
}

// ---------------------------------------------------------------------------
extern "C" void kernel_launch(void* const* d_in, const int* in_sizes, int n_in,
                              void* d_out, int out_size, void* d_ws, size_t ws_size,
                              hipStream_t stream)
{
  const float* x    = (const float*)d_in[0];
  const int*   eix  = (const int*)d_in[1];
  const float* ea   = (const float*)d_in[2];
  const float* WA   = (const float*)d_in[3];
  const float* bA   = (const float*)d_in[4];
  const float* WB   = (const float*)d_in[5];
  const float* bB   = (const float*)d_in[6];
  const float* WC   = (const float*)d_in[7];
  const float* bC   = (const float*)d_in[8];
  const float* WD   = (const float*)d_in[9];
  const float* bD   = (const float*)d_in[10];
  const float* WE   = (const float*)d_in[11];
  const float* bE   = (const float*)d_in[12];
  const float* bnxg = (const float*)d_in[13];
  const float* bnxb = (const float*)d_in[14];
  const float* bneg = (const float*)d_in[15];
  const float* bneb = (const float*)d_in[16];
  const float* n1g  = (const float*)d_in[17];
  const float* n1b  = (const float*)d_in[18];
  const float* n2g  = (const float*)d_in[19];
  const float* n2b  = (const float*)d_in[20];
  const float* Wf1  = (const float*)d_in[21];
  const float* bf1  = (const float*)d_in[22];
  const float* Wf2  = (const float*)d_in[23];
  const float* bf2  = (const float*)d_in[24];

  const int N = in_sizes[0] / DD;
  const int E = in_sizes[2] / DD;
  const long long ND = (long long)N * DD;
  const long long ED = (long long)E * DD;

  float* ws  = (float*)d_ws;
  float* Ax  = ws;               // [N,D]  -> reused as h_bn
  float* Bx  = ws + ND;          // [N,D]  -> reused as x_final
  float* Dx  = ws + 2 * ND;      // [N,D]  \ reused as h1 [N,2D]
  float* Ex  = ws + 3 * ND;      // [N,D]  /
  float* eij = ws + 4 * ND;      // [E,D]  Ce -> e_ij -> (first ND) h2
  float* agg = ws + 4 * ND + ED; // [N,D]  aggr -> x_trans -> y
  float* st  = ws + 5 * ND + ED; // stats: 4x(sum,sumsq)=1024 + scale/shift=1024
  float* st_x = st, *st_e = st + 256, *st_1 = st + 512, *st_2 = st + 768;
  float* sc_x = st + 1024, *sh_x = st + 1152;
  float* sc_e = st + 1280, *sh_e = st + 1408;
  float* sc_1 = st + 1536, *sh_1 = st + 1664;
  float* sc_2 = st + 1792, *sh_2 = st + 1920;

  float* out_x = (float*)d_out;         // [N,D]
  float* out_e = (float*)d_out + ND;    // [E,D]

  hipMemsetAsync(agg, 0, ND * sizeof(float), stream);
  hipMemsetAsync(st, 0, 1024 * sizeof(float), stream);

  dim3 blk(256);
  dim3 gN((N + BM - 1) / BM, DD / BN);       // node GEMMs: 391 x 2
  dim3 gE((E + BM - 1) / BM, DD / BN);       // edge GEMM: 6250 x 2
  dim3 gF1((N + BM - 1) / BM, 2 * DD / BN);  // FFN1: 391 x 4

  // 5 projection GEMMs (bf16 WMMA, fp32 accumulate)
  wmma_gemm_bias<<<gN, blk, 0, stream>>>(x, WA, bA, Ax, N, DD, DD, 0);
  wmma_gemm_bias<<<gN, blk, 0, stream>>>(x, WB, bB, Bx, N, DD, DD, 0);
  wmma_gemm_bias<<<gN, blk, 0, stream>>>(x, WD, bD, Dx, N, DD, DD, 0);
  wmma_gemm_bias<<<gN, blk, 0, stream>>>(x, WE, bE, Ex, N, DD, DD, 0);
  wmma_gemm_bias<<<gE, blk, 0, stream>>>(ea, WC, bC, eij, E, DD, DD, 0);

  // gather -> gate -> scatter-add (memory-bound core) + edge BN stats
  edge_gate_scatter<<<1024, 256, 0, stream>>>(eix, Dx, Ex, Bx, eij, agg, st_e, E);

  // x_trans = Ax + aggr, node BN stats
  node_xtrans<<<512, 256, 0, stream>>>(Ax, agg, st_x, N);

  bn_finalize<<<1, 128, 0, stream>>>(st_x, bnxg, bnxb, 1.f / (float)N, sc_x, sh_x);
  bn_finalize<<<1, 128, 0, stream>>>(st_e, bneg, bneb, 1.f / (float)E, sc_e, sh_e);

  // residuals + relu(bn(.)); x path also accumulates n1 stats
  apply_x<<<512, 256, 0, stream>>>(x, agg, sc_x, sh_x, Bx /*x_final*/, st_1, N);
  apply_e<<<1024, 256, 0, stream>>>(ea, eij, sc_e, sh_e, out_e, E);

  // FFN block: h = bn(x_final,n1); h = relu(h@Wf1+bf1)@Wf2+bf2
  bn_finalize<<<1, 128, 0, stream>>>(st_1, n1g, n1b, 1.f / (float)N, sc_1, sh_1);
  affine_cols<<<512, 256, 0, stream>>>(Bx, sc_1, sh_1, Ax /*h_bn*/, N);
  wmma_gemm_bias<<<gF1, blk, 0, stream>>>(Ax, Wf1, bf1, Dx /*h1 [N,2D]*/, N, DD, 2 * DD, 1);
  wmma_gemm_bias<<<gN, blk, 0, stream>>>(Dx, Wf2, bf2, eij /*h2 [N,D]*/, N, 2 * DD, DD, 0);

  // y = x_final + h2 ; n2 stats ; out_x = bn(y, n2)
  add_stats<<<512, 256, 0, stream>>>(Bx, eij, agg /*y*/, st_2, N);
  bn_finalize<<<1, 128, 0, stream>>>(st_2, n2g, n2b, 1.f / (float)N, sc_2, sh_2);
  affine_cols<<<512, 256, 0, stream>>>(agg, sc_2, sh_2, out_x, N);
}
